// MultiHeadAttention_83468394430674
// MI455X (gfx1250) — matrix operand order repaired and checked by reading, not compiled, not run on hardware
//
#include <hip/hip_runtime.h>

#define D_MODEL 768
#define N_HEADS 12
#define HEAD    64
#define BATCH   2
#define SEQ     2048
#define M_TOTAL (BATCH*SEQ)   // 4096
#define QKV_N   (3*D_MODEL)   // 2304

typedef __attribute__((ext_vector_type(16))) __bf16 v16bf;
typedef __attribute__((ext_vector_type(8)))  float  v8f;

union FragAB { unsigned int u[8]; v16bf v; };
union FragC  { float f[8];        v8f   v; };

__device__ __forceinline__ unsigned short f2bf(float f) {
    unsigned int u = __float_as_uint(f);
    unsigned int r = u + 0x7FFFu + ((u >> 16) & 1u);   // round-to-nearest-even
    return (unsigned short)(r >> 16);
}

__global__ void cvt_f32_bf16(const float* __restrict__ in,
                             unsigned short* __restrict__ out, int n) {
    int i = blockIdx.x * blockDim.x + threadIdx.x;
    if (i < n) out[i] = f2bf(in[i]);
}

// ---------------------------------------------------------------------------
// QKV projection: qkv[m,n] = sum_k X[m,k] * W1[n,k] + b1[n]
// Block tile 128x128, 8 waves, wave tile 32x64, K-step 32 (bf16 WMMA).
// Output scattered into Q/K/V in (B,H,T,Dh) bf16 layout.
// ---------------------------------------------------------------------------
__global__ __launch_bounds__(256)
void qkv_gemm(const unsigned short* __restrict__ Xbf,
              const unsigned short* __restrict__ W1bf,
              const float* __restrict__ b1,
              unsigned short* __restrict__ Qb,
              unsigned short* __restrict__ Kb,
              unsigned short* __restrict__ Vb) {
    __shared__ __align__(16) unsigned short As[128 * 34];
    __shared__ __align__(16) unsigned short Bs[128 * 34];
    const int tid  = threadIdx.x;
    const int lane = tid & 31, wave = tid >> 5;
    const int half = lane >> 4, l16 = lane & 15;
    const int mBase = blockIdx.y * 128;
    const int nBase = blockIdx.x * 128;
    const int waveM = (wave >> 1) * 32;
    const int waveN = (wave & 1) * 64;

    FragC acc[2][4];
    #pragma unroll
    for (int s = 0; s < 2; ++s)
        #pragma unroll
        for (int t = 0; t < 4; ++t)
            #pragma unroll
            for (int r = 0; r < 8; ++r) acc[s][t].f[r] = 0.f;

    const unsigned int* As32 = (const unsigned int*)As;
    const unsigned int* Bs32 = (const unsigned int*)Bs;
    const int srow = tid >> 1;        // 0..127
    const int scol = (tid & 1) * 16;  // 0 or 16

    for (int k0 = 0; k0 < D_MODEL; k0 += 32) {
        {   // stage X tile (row-major, stride 34 for bank-conflict-free reads)
            const uint4* gp = (const uint4*)(Xbf + (size_t)(mBase + srow) * D_MODEL + k0 + scol);
            uint4 d0 = gp[0], d1 = gp[1];
            unsigned int* dst = (unsigned int*)(As + srow * 34 + scol);
            dst[0]=d0.x; dst[1]=d0.y; dst[2]=d0.z; dst[3]=d0.w;
            dst[4]=d1.x; dst[5]=d1.y; dst[6]=d1.z; dst[7]=d1.w;
        }
        {   // stage W1 tile (n-major == B fragment friendly)
            const uint4* gp = (const uint4*)(W1bf + (size_t)(nBase + srow) * D_MODEL + k0 + scol);
            uint4 d0 = gp[0], d1 = gp[1];
            unsigned int* dst = (unsigned int*)(Bs + srow * 34 + scol);
            dst[0]=d0.x; dst[1]=d0.y; dst[2]=d0.z; dst[3]=d0.w;
            dst[4]=d1.x; dst[5]=d1.y; dst[6]=d1.z; dst[7]=d1.w;
        }
        __syncthreads();

        FragAB a[2], b[4];
        #pragma unroll
        for (int s = 0; s < 2; ++s) {
            int row = waveM + s * 16 + l16;
            #pragma unroll
            for (int i = 0; i < 8; ++i) {
                int kp = (i >> 2) * 16 + (i & 3) * 2 + half * 8;  // even
                a[s].u[i] = As32[row * 17 + (kp >> 1)];
            }
        }
        #pragma unroll
        for (int t = 0; t < 4; ++t) {
            int col = waveN + t * 16 + l16;
            #pragma unroll
            for (int i = 0; i < 8; ++i)
                b[t].u[i] = Bs32[col * 17 + (i >> 2) * 8 + half * 4 + (i & 3)];
        }
        #pragma unroll
        for (int s = 0; s < 2; ++s)
            #pragma unroll
            for (int t = 0; t < 4; ++t)
                acc[s][t].v = __builtin_amdgcn_wmma_f32_16x16x32_bf16(
                    false, a[s].v, false, b[t].v, (short)0, acc[s][t].v, false, false);
        __syncthreads();
    }

    // epilogue: bias + scatter to Q/K/V (B,H,T,Dh) bf16
    #pragma unroll
    for (int s = 0; s < 2; ++s) {
        #pragma unroll
        for (int t = 0; t < 4; ++t) {
            int n = nBase + waveN + t * 16 + l16;
            float bias = b1[n];
            int which = n / D_MODEL;
            int rem   = n - which * D_MODEL;
            int h = rem / HEAD, d = rem - (rem / HEAD) * HEAD;
            unsigned short* dst = (which == 0) ? Qb : (which == 1) ? Kb : Vb;
            #pragma unroll
            for (int r = 0; r < 8; ++r) {
                int m  = mBase + waveM + s * 16 + r + half * 8;
                int bb = m >> 11, tt = m & (SEQ - 1);
                dst[(((size_t)(bb * N_HEADS + h) * SEQ + tt) * HEAD) + d] =
                    f2bf(acc[s][t].f[r] + bias);
            }
        }
    }
}

// ---------------------------------------------------------------------------
// Flash attention: block = (b, h, 128 q-rows), 8 waves x 16 q-rows each.
// K/V tiles of 32 keys staged in LDS; online softmax; O accumulated in f32.
// ---------------------------------------------------------------------------
__global__ __launch_bounds__(256)
void attn(const unsigned short* __restrict__ Qb,
          const unsigned short* __restrict__ Kb,
          const unsigned short* __restrict__ Vb,
          unsigned short* __restrict__ Obf) {
    __shared__ __align__(16) unsigned short Ks[32 * 68];   // [key][dh]
    __shared__ __align__(16) unsigned short Vst[64 * 36];  // [dh][key] (transposed)
    __shared__ __align__(16) float          Sw[8][16 * 33];
    __shared__ __align__(16) unsigned short Pw[8][16 * 34];
    __shared__ float alphaW[8][16];

    const int tid  = threadIdx.x;
    const int lane = tid & 31, wave = tid >> 5;
    const int half = lane >> 4, l16 = lane & 15;
    const int qBase = blockIdx.x * 128;
    const int h = blockIdx.y, b = blockIdx.z;
    const size_t headOff = (size_t)(b * N_HEADS + h) * SEQ * HEAD;
    const unsigned short* Qh = Qb + headOff;
    const unsigned short* Kh = Kb + headOff;
    const unsigned short* Vh = Vb + headOff;

    // Q fragments for this wave's 16 rows (kept in registers for all KV tiles)
    FragAB qf[2];
    {
        int row = qBase + wave * 16 + l16;
        const unsigned int* Q32 = (const unsigned int*)Qh;
        #pragma unroll
        for (int f = 0; f < 2; ++f)
            #pragma unroll
            for (int i = 0; i < 8; ++i) {
                int kp = f * 32 + (i >> 2) * 16 + (i & 3) * 2 + half * 8;
                qf[f].u[i] = Q32[row * 32 + (kp >> 1)];
            }
    }

    FragC oacc[4];
    #pragma unroll
    for (int f = 0; f < 4; ++f)
        #pragma unroll
        for (int r = 0; r < 8; ++r) oacc[f].f[r] = 0.f;

    float mrow = -3.0e38f, lrow = 0.f;

    const unsigned int* Ks32  = (const unsigned int*)Ks;
    const unsigned int* Vst32 = (const unsigned int*)Vst;
    float*          Sme   = Sw[wave];
    unsigned short* Pme   = Pw[wave];
    const unsigned int* Pme32 = (const unsigned int*)Pme;

    const int nTiles = (qBase >> 5) + 4;   // causal limit for this q-block
    for (int kt = 0; kt < nTiles; ++kt) {
        const int keyBase = kt * 32;
        {   // stage K tile (32 keys x 64 dh)
            int row = tid >> 3, col = (tid & 7) * 8;
            const uint4* gp = (const uint4*)(Kh + (size_t)(keyBase + row) * HEAD + col);
            uint4 d = *gp;
            unsigned int* dst = (unsigned int*)(Ks + row * 68 + col);
            dst[0]=d.x; dst[1]=d.y; dst[2]=d.z; dst[3]=d.w;
        }
        {   // stage V tile transposed: Vst[dh][key]
            int key = tid >> 3, dh0 = (tid & 7) * 8;
            const uint4* gp = (const uint4*)(Vh + (size_t)(keyBase + key) * HEAD + dh0);
            uint4 d = *gp;
            unsigned int w[4] = {d.x, d.y, d.z, d.w};
            #pragma unroll
            for (int j = 0; j < 4; ++j) {
                Vst[(dh0 + 2*j    ) * 36 + key] = (unsigned short)(w[j] & 0xffffu);
                Vst[(dh0 + 2*j + 1) * 36 + key] = (unsigned short)(w[j] >> 16);
            }
        }
        __syncthreads();

        // S = Q K^T  (16 x 32 per wave, two K-steps over Dh)
        FragC sacc[2];
        #pragma unroll
        for (int t = 0; t < 2; ++t)
            #pragma unroll
            for (int r = 0; r < 8; ++r) sacc[t].f[r] = 0.f;
        #pragma unroll
        for (int ks = 0; ks < 2; ++ks) {
            FragAB kf[2];
            #pragma unroll
            for (int t = 0; t < 2; ++t) {
                int key = t * 16 + l16;
                #pragma unroll
                for (int i = 0; i < 8; ++i)
                    kf[t].u[i] = Ks32[key * 34 + ks * 16 + (i >> 2) * 8 + half * 4 + (i & 3)];
            }
            #pragma unroll
            for (int t = 0; t < 2; ++t)
                sacc[t].v = __builtin_amdgcn_wmma_f32_16x16x32_bf16(
                    false, qf[ks].v, false, kf[t].v, (short)0, sacc[t].v, false, false);
        }

        // scale + causal mask, spill S to per-wave LDS
        #pragma unroll
        for (int t = 0; t < 2; ++t)
            #pragma unroll
            for (int r = 0; r < 8; ++r) {
                int qRow = qBase + wave * 16 + r + half * 8;
                int kCol = keyBase + t * 16 + l16;
                float v = (kCol <= qRow) ? sacc[t].f[r] * 0.125f : -3.0e38f;
                Sme[(r + half * 8) * 33 + t * 16 + l16] = v;
            }
        asm volatile("s_wait_dscnt 0" ::: "memory");   // wave-internal LDS handoff

        // online softmax: lane r < 16 owns query row r
        if (lane < 16) {
            const float* srow = &Sme[lane * 33];
            float tmax = srow[0];
            #pragma unroll
            for (int j = 1; j < 32; ++j) tmax = fmaxf(tmax, srow[j]);
            float mnew  = fmaxf(mrow, tmax);
            float alpha = __expf(mrow - mnew);
            float psum  = 0.f;
            #pragma unroll
            for (int j = 0; j < 32; ++j) {
                float p = __expf(srow[j] - mnew);
                psum += p;
                Pme[lane * 34 + j] = f2bf(p);
            }
            lrow = alpha * lrow + psum;
            mrow = mnew;
            alphaW[wave][lane] = alpha;
        }
        asm volatile("s_wait_dscnt 0" ::: "memory");

        // rescale O accumulators by alpha(row)
        #pragma unroll
        for (int f = 0; f < 4; ++f)
            #pragma unroll
            for (int r = 0; r < 8; ++r)
                oacc[f].f[r] *= alphaW[wave][r + half * 8];

        // O += P V : P is A-fragment (16x32), V B-fragments from transposed LDS
        FragAB pf;
        #pragma unroll
        for (int i = 0; i < 8; ++i) {
            int kp = (i >> 2) * 16 + (i & 3) * 2 + half * 8;
            pf.u[i] = Pme32[l16 * 17 + (kp >> 1)];
        }
        #pragma unroll
        for (int f = 0; f < 4; ++f) {
            FragAB vf;
            int dh = f * 16 + l16;
            #pragma unroll
            for (int i = 0; i < 8; ++i)
                vf.u[i] = Vst32[dh * 18 + (i >> 2) * 8 + half * 4 + (i & 3)];
            oacc[f].v = __builtin_amdgcn_wmma_f32_16x16x32_bf16(
                false, pf.v, false, vf.v, (short)0, oacc[f].v, false, false);
        }
        __syncthreads();
    }

    // finalize: O /= l, store bf16 into (B,T,C) layout for the output GEMM
    if (lane < 16) alphaW[wave][lane] = 1.0f / lrow;
    asm volatile("s_wait_dscnt 0" ::: "memory");
    #pragma unroll
    for (int f = 0; f < 4; ++f)
        #pragma unroll
        for (int r = 0; r < 8; ++r) {
            int qRow = qBase + wave * 16 + r + half * 8;
            float val = oacc[f].f[r] * alphaW[wave][r + half * 8];
            int dh = f * 16 + l16;
            Obf[(size_t)(b * SEQ + qRow) * D_MODEL + h * HEAD + dh] = f2bf(val);
        }
}

// ---------------------------------------------------------------------------
// Output projection: out[m,n] = sum_c O[m,c] * W2[n,c] + b2[n]  (fp32 out)
// ---------------------------------------------------------------------------
__global__ __launch_bounds__(256)
void proj_gemm(const unsigned short* __restrict__ Obf,
               const unsigned short* __restrict__ W2bf,
               const float* __restrict__ b2,
               float* __restrict__ out) {
    __shared__ __align__(16) unsigned short As[128 * 34];
    __shared__ __align__(16) unsigned short Bs[128 * 34];
    const int tid  = threadIdx.x;
    const int lane = tid & 31, wave = tid >> 5;
    const int half = lane >> 4, l16 = lane & 15;
    const int mBase = blockIdx.y * 128;
    const int nBase = blockIdx.x * 128;
    const int waveM = (wave >> 1) * 32;
    const int waveN = (wave & 1) * 64;

    FragC acc[2][4];
    #pragma unroll
    for (int s = 0; s < 2; ++s)
        #pragma unroll
        for (int t = 0; t < 4; ++t)
            #pragma unroll
            for (int r = 0; r < 8; ++r) acc[s][t].f[r] = 0.f;

    const unsigned int* As32 = (const unsigned int*)As;
    const unsigned int* Bs32 = (const unsigned int*)Bs;
    const int srow = tid >> 1;
    const int scol = (tid & 1) * 16;

    for (int k0 = 0; k0 < D_MODEL; k0 += 32) {
        {
            const uint4* gp = (const uint4*)(Obf + (size_t)(mBase + srow) * D_MODEL + k0 + scol);
            uint4 d0 = gp[0], d1 = gp[1];
            unsigned int* dst = (unsigned int*)(As + srow * 34 + scol);
            dst[0]=d0.x; dst[1]=d0.y; dst[2]=d0.z; dst[3]=d0.w;
            dst[4]=d1.x; dst[5]=d1.y; dst[6]=d1.z; dst[7]=d1.w;
        }
        {
            const uint4* gp = (const uint4*)(W2bf + (size_t)(nBase + srow) * D_MODEL + k0 + scol);
            uint4 d0 = gp[0], d1 = gp[1];
            unsigned int* dst = (unsigned int*)(Bs + srow * 34 + scol);
            dst[0]=d0.x; dst[1]=d0.y; dst[2]=d0.z; dst[3]=d0.w;
            dst[4]=d1.x; dst[5]=d1.y; dst[6]=d1.z; dst[7]=d1.w;
        }
        __syncthreads();

        FragAB a[2], b[4];
        #pragma unroll
        for (int s = 0; s < 2; ++s) {
            int row = waveM + s * 16 + l16;
            #pragma unroll
            for (int i = 0; i < 8; ++i) {
                int kp = (i >> 2) * 16 + (i & 3) * 2 + half * 8;
                a[s].u[i] = As32[row * 17 + (kp >> 1)];
            }
        }
        #pragma unroll
        for (int t = 0; t < 4; ++t) {
            int col = waveN + t * 16 + l16;
            #pragma unroll
            for (int i = 0; i < 8; ++i)
                b[t].u[i] = Bs32[col * 17 + (i >> 2) * 8 + half * 4 + (i & 3)];
        }
        #pragma unroll
        for (int s = 0; s < 2; ++s)
            #pragma unroll
            for (int t = 0; t < 4; ++t)
                acc[s][t].v = __builtin_amdgcn_wmma_f32_16x16x32_bf16(
                    false, a[s].v, false, b[t].v, (short)0, acc[s][t].v, false, false);
        __syncthreads();
    }

    #pragma unroll
    for (int s = 0; s < 2; ++s)
        #pragma unroll
        for (int t = 0; t < 4; ++t) {
            int n = nBase + waveN + t * 16 + l16;
            float bias = b2[n];
            #pragma unroll
            for (int r = 0; r < 8; ++r) {
                int m = mBase + waveM + s * 16 + r + half * 8;
                out[(size_t)m * D_MODEL + n] = acc[s][t].f[r] + bias;
            }
        }
}

// ---------------------------------------------------------------------------
extern "C" void kernel_launch(void* const* d_in, const int* in_sizes, int n_in,
                              void* d_out, int out_size, void* d_ws, size_t ws_size,
                              hipStream_t stream) {
    const float* x  = (const float*)d_in[0];
    const float* W1 = (const float*)d_in[1];
    const float* b1 = (const float*)d_in[2];
    const float* W2 = (const float*)d_in[3];
    const float* b2 = (const float*)d_in[4];
    float* out = (float*)d_out;

    unsigned short* ws = (unsigned short*)d_ws;
    size_t o = 0;
    unsigned short* Xbf  = ws + o; o += (size_t)M_TOTAL * D_MODEL;
    unsigned short* W1bf = ws + o; o += (size_t)QKV_N * D_MODEL;
    unsigned short* W2bf = ws + o; o += (size_t)D_MODEL * D_MODEL;
    unsigned short* Qb   = ws + o; o += (size_t)M_TOTAL * D_MODEL;
    unsigned short* Kb   = ws + o; o += (size_t)M_TOTAL * D_MODEL;
    unsigned short* Vb   = ws + o; o += (size_t)M_TOTAL * D_MODEL;
    unsigned short* Obf  = ws + o; o += (size_t)M_TOTAL * D_MODEL;

    const int nX  = M_TOTAL * D_MODEL;
    const int nW1 = QKV_N * D_MODEL;
    const int nW2 = D_MODEL * D_MODEL;
    cvt_f32_bf16<<<(nX  + 255) / 256, 256, 0, stream>>>(x,  Xbf,  nX);
    cvt_f32_bf16<<<(nW1 + 255) / 256, 256, 0, stream>>>(W1, W1bf, nW1);
    cvt_f32_bf16<<<(nW2 + 255) / 256, 256, 0, stream>>>(W2, W2bf, nW2);

    qkv_gemm<<<dim3(QKV_N / 128, M_TOTAL / 128), 256, 0, stream>>>(Xbf, W1bf, b1, Qb, Kb, Vb);
    attn<<<dim3(SEQ / 128, N_HEADS, BATCH), 256, 0, stream>>>(Qb, Kb, Vb, Obf);
    proj_gemm<<<dim3(D_MODEL / 128, M_TOTAL / 128), 256, 0, stream>>>(Obf, W2bf, b2, out);
}
